// TritonPackedBitLinear_47141561041543
// MI455X (gfx1250) — matrix operand order repaired and checked by reading, not compiled, not run on hardware
//
#include <hip/hip_runtime.h>

// ---------------------------------------------------------------------------
// 1-bit-sign GEMV (B=1): out[n] = sum_k sign[n,k] * scale[n, k/128] * x[k]
//   N = 32768, K = 8192, group = 128. Harness dtypes: x/scales -> f32,
//   packed_signs -> one int32 per packed byte, out -> f32.
//
// Roofline: signs dominate at 128 MB -> ~5.9 us @ 23.3 TB/s. The kernel is
// memory-bound *if* the bit->(+-1) unpack is free, so:
//   * MACs on the WMMA pipe: v_wmma_f32_16x16x32_f16, B = sign matrix from a
//     256-entry LDS LUT (byte -> 8 x f16 +-1, one ds_load_b128 per byte),
//     A = x chunk replicated over the wasted M dim (B=1 GEMV).
//   * Sign stream (the 128 MB) staged by the Tensor Data Mover:
//     2D tile 128 rows x 64 ints, row stride 1024 ints, double-buffered in
//     LDS, TENSORcnt-gated. pad_enable inserts 16B per 256B row during the
//     DMA -> 272B LDS pitch -> conflict-free ds_load_b64 sign reads
//     (16 rows x 4-bank step x lo/hi halves = all 64 banks exactly once).
// ---------------------------------------------------------------------------

#define N_OUT 32768
#define K_IN  8192
#define GS    128
#define G_N   (K_IN / GS)          // 64 scale groups per row
#define KP    (K_IN / 8)           // 1024 packed "bytes" (int32 each) per row

#define ROWS_PER_BLOCK 128
#define CG         4               // scale groups per staged chunk
#define CH         (G_N / CG)      // 16 chunks
#define CHUNK_INTS (CG * (GS / 8)) // 64 ints per row per chunk
#define ROW_PITCH  (CHUNK_INTS + 4) // 68 ints: +16B pad per row (bank spread)

typedef _Float16 half8  __attribute__((ext_vector_type(8)));
typedef _Float16 half16 __attribute__((ext_vector_type(16)));
typedef float    float8 __attribute__((ext_vector_type(8)));
typedef unsigned int uint4v __attribute__((ext_vector_type(4)));
typedef int      int8v  __attribute__((ext_vector_type(8)));
typedef int      int4v  __attribute__((ext_vector_type(4)));

#if __has_builtin(__builtin_amdgcn_tensor_load_to_lds)
#define USE_TDM 1
#else
#define USE_TDM 0
#endif

#if USE_TDM
// Issue one TDM 2D tile load: 128 rows x 64 ints, row stride 1024 ints,
// destination rows padded to 272B. Descriptor per CDNA5 ISA.
__device__ __forceinline__ void tdm_load_chunk(unsigned lds_base,
                                               unsigned long long gaddr)
{
    uint4v g0;
    g0.x = 1u;                                         // count=1 valid user D#
    g0.y = lds_base;                                   // lds_addr (bytes)
    g0.z = (unsigned)gaddr;                            // global_addr[31:0]
    g0.w = (unsigned)((gaddr >> 32) & 0x01FFFFFFull)   // global_addr[56:32]
         | 0x80000000u;                                // type = 2 ("image")

    int8v g1;
    g1[0] = (int)((2u << 16)      // data_size = 4 bytes
                | (1u << 20)      // pad_enable
                | (5u << 22)      // pad_interval: 64 dwords (one tile row)
                | (3u << 25));    // pad_amount: 4 dwords (16 bytes)
    g1[1] = (int)(((unsigned)KP & 0xFFFFu) << 16);     // tensor_dim0 lo16
    g1[2] = (int)(((unsigned)KP >> 16)                 // tensor_dim0 hi16
                | (((unsigned)N_OUT & 0xFFFFu) << 16));// tensor_dim1 lo16
    g1[3] = (int)(((unsigned)N_OUT >> 16)              // tensor_dim1 hi16
                | ((unsigned)CHUNK_INTS << 16));       // tile_dim0 = 64
    g1[4] = (int)((unsigned)ROWS_PER_BLOCK);           // tile_dim1=128, dim2=0
    g1[5] = (int)((unsigned)KP);                       // tensor_dim0_stride lo
    g1[6] = 0;                                         // stride hi | dim1_stride
    g1[7] = 0;

    int4v z4 = {0, 0, 0, 0};
#if defined(__clang_major__) && (__clang_major__ >= 23)
    int8v z8 = {0, 0, 0, 0, 0, 0, 0, 0};
    __builtin_amdgcn_tensor_load_to_lds(g0, g1, z4, z4, z8, 0);
#else
    __builtin_amdgcn_tensor_load_to_lds(g0, g1, z4, z4, 0);
#endif
}
#endif

__global__ __launch_bounds__(256)
void bitlinear_wmma_kernel(const float* __restrict__ x,
                           const int*   __restrict__ signs,
                           const float* __restrict__ scales,
                           float*       __restrict__ out)
{
    // 4 KB sign LUT: entry v, slot i = (bit i of v) ? +1.0h : -1.0h
    __shared__ __attribute__((aligned(16))) _Float16 lutS[256][8];
    // 16 KB staged activation vector (f32 -> f16 once per block)
    __shared__ __attribute__((aligned(16))) _Float16 xS[K_IN];
#if USE_TDM
    // 2 x 34 KB double-buffered sign stage (TDM destination, padded rows)
    __shared__ __attribute__((aligned(16))) int stageS[2][ROWS_PER_BLOCK * ROW_PITCH];
#endif

    const int tid  = threadIdx.x;
    const int wave = tid >> 5;        // 8 waves per block
    const int lane = tid & 31;        // wave32
    const int col  = lane & 15;       // WMMA N column (output row within tile)
    const int hi   = lane >> 4;       // 0: low-K half-wave, 1: high-K half-wave

    // ---- build LUT (256 threads -> 256 entries, one ds_store_b128 each) ----
    {
        half8 e;
        #pragma unroll
        for (int i = 0; i < 8; ++i)
            e[i] = ((tid >> i) & 1) ? (_Float16)1.0f : (_Float16)(-1.0f);
        *(half8*)&lutS[tid][0] = e;
    }
    // ---- stage x into LDS as f16 (32 elements per thread) ----
    #pragma unroll 4
    for (int j = 0; j < K_IN / 256; ++j) {
        const int idx = tid * (K_IN / 256) + j;
        xS[idx] = (_Float16)x[idx];
    }
    __syncthreads();

    const int    n0    = blockIdx.x * ROWS_PER_BLOCK + wave * 16;
    const int    row   = n0 + col;                      // this lane's N column
    const float* scrow = scales + (size_t)row * G_N;
    float acc = 0.0f;

#if USE_TDM
    const unsigned long long gbase = (unsigned long long)(uintptr_t)signs
        + (unsigned long long)blockIdx.x * ROWS_PER_BLOCK * KP * 4ull;
    if (wave == 0)
        tdm_load_chunk((unsigned)(uintptr_t)&stageS[0][0], gbase);
    const int rbase = (wave * 16 + col) * ROW_PITCH;    // stage row (ints)
#else
    const int* srow = signs + (size_t)row * KP;
#endif

    for (int ch = 0; ch < CH; ++ch) {
#if USE_TDM
        // All waves done reading the buffer the next DMA will overwrite.
        __syncthreads();
        if (wave == 0) {
            if (ch + 1 < CH) {
                tdm_load_chunk((unsigned)(uintptr_t)&stageS[(ch + 1) & 1][0],
                               gbase + (unsigned long long)(ch + 1) * CHUNK_INTS * 4ull);
                __builtin_amdgcn_s_wait_tensorcnt(1);   // chunk ch landed
            } else {
                __builtin_amdgcn_s_wait_tensorcnt(0);
            }
        }
        __syncthreads();                                 // publish stage[ch&1]
        const int* stg = &stageS[ch & 1][0];
#endif
        #pragma unroll
        for (int cg = 0; cg < CG; ++cg) {
            const int g = ch * CG + cg;
            float8 c = {0.f, 0.f, 0.f, 0.f, 0.f, 0.f, 0.f, 0.f};

            #pragma unroll
            for (int t = 0; t < 4; ++t) {
                const int kc = g * GS + t * 32;   // K chunk base (32 wide)

                // A (16x32 f16): x chunk, identical in every M row.
                const half8  alo = *(const half8*)&xS[kc + hi * 16];
                const half8  ahi = *(const half8*)&xS[kc + hi * 16 + 8];
                const half16 A   = __builtin_shufflevector(alo, ahi,
                                     0,1,2,3,4,5,6,7,8,9,10,11,12,13,14,15);

                // B (32x16 f16): two packed bytes -> two LUT b128 gathers.
#if USE_TDM
                const int2 pb = *(const int2*)&stg[rbase + cg * 16 + t * 4 + hi * 2];
#else
                const int2 pb = *(const int2*)(srow + g * 16 + t * 4 + hi * 2);
#endif
                const half8  blo = *(const half8*)&lutS[pb.x & 255][0];
                const half8  bhi = *(const half8*)&lutS[pb.y & 255][0];
                const half16 B   = __builtin_shufflevector(blo, bhi,
                                     0,1,2,3,4,5,6,7,8,9,10,11,12,13,14,15);

                c = __builtin_amdgcn_wmma_f32_16x16x32_f16(
                        /*neg_a=*/false, A, /*neg_b=*/false, B,
                        /*c_mod=*/(short)0, c, /*reuse_a=*/false, /*reuse_b=*/false);
            }
            // All M rows equal -> c[0] is this column's group dot on every lane.
            acc = __builtin_fmaf(c[0], scrow[g], acc);
        }
    }

    if (lane < 16)
        out[row] = acc;
}

extern "C" void kernel_launch(void* const* d_in, const int* in_sizes, int n_in,
                              void* d_out, int out_size, void* d_ws, size_t ws_size,
                              hipStream_t stream) {
    const float* x      = (const float*)d_in[0];
    const int*   signs  = (const int*)  d_in[1];
    const float* scales = (const float*)d_in[2];
    float*       out    = (float*)d_out;

    dim3 block(256);
    dim3 grid(N_OUT / ROWS_PER_BLOCK);   // 256 workgroups, 16 rows per wave
    hipLaunchKernelGGL(bitlinear_wmma_kernel, grid, block, 0, stream,
                       x, signs, scales, out);

    (void)in_sizes; (void)n_in; (void)out_size; (void)d_ws; (void)ws_size;
}
